// RecurrentNeuralNetwork_31550829757085
// MI455X (gfx1250) — matrix operand order repaired and checked by reading, not compiled, not run on hardware
//
#include <hip/hip_runtime.h>

#define NI 128
#define NH 1024
#define NO 64
#define BB 128
#define TT 256
#define ALPHA_C    0.25f
#define NOISE_STD  0.025f   // sigma_neu * sqrt(alpha) = 0.05 * 0.5

typedef __attribute__((ext_vector_type(16))) __bf16        v16bf;
typedef __attribute__((ext_vector_type(2)))  __bf16        v2bf;
typedef __attribute__((ext_vector_type(8)))  float         v8f;
typedef __attribute__((ext_vector_type(4)))  unsigned int  u32x4;

union Frag16 { v16bf v; u32x4 q[2]; unsigned int u[8]; unsigned short s[16]; };
union FragC  { v8f   v; float f[8]; };

#if __has_builtin(__builtin_amdgcn_tanhf)
#define TANHF(x) __builtin_amdgcn_tanhf(x)
#else
#define TANHF(x) tanhf(x)
#endif

// Native bf16 conversion: backend lowers to v_cvt_pk_bf16_f32 on gfx1250.
__device__ __forceinline__ unsigned int pack2bf(float lo, float hi) {
  v2bf p;
  p[0] = (__bf16)lo;
  p[1] = (__bf16)hi;
  return __builtin_bit_cast(unsigned int, p);
}
__device__ __forceinline__ unsigned short f2bf(float x) {
  __bf16 b = (__bf16)x;
  return __builtin_bit_cast(unsigned short, b);
}

// Device-wide split barrier: monotonically increasing counter, no reset.
__device__ __forceinline__ void gridbar(unsigned int* ctr, unsigned int target) {
  __syncthreads();
  if (threadIdx.x == 0) {
    __threadfence();  // release prior writes to agent scope
    __hip_atomic_fetch_add(ctr, 1u, __ATOMIC_RELEASE, __HIP_MEMORY_SCOPE_AGENT);
    while (__hip_atomic_load(ctr, __ATOMIC_ACQUIRE, __HIP_MEMORY_SCOPE_AGENT) < target) {
      __builtin_amdgcn_s_sleep(2);
    }
  }
  __syncthreads();
}

// Fill an A-style fragment from fp32 memory: lanes<16 take K{0..7,16..23},
// lanes>=16 take K{8..15,24..31} (ks pre-offset by caller).
__device__ __forceinline__ void loadA_f32(Frag16& a, const float* __restrict__ p) {
#pragma unroll
  for (int i = 0; i < 4; ++i) a.u[i]     = pack2bf(p[2 * i],      p[2 * i + 1]);
#pragma unroll
  for (int i = 0; i < 4; ++i) a.u[4 + i] = pack2bf(p[16 + 2 * i], p[16 + 2 * i + 1]);
}
// B-style fragment: 16 contiguous K values (kb pre-offset by caller).
__device__ __forceinline__ void loadB_f32(Frag16& b, const float* __restrict__ p) {
#pragma unroll
  for (int i = 0; i < 8; ++i) b.u[i] = pack2bf(p[2 * i], p[2 * i + 1]);
}

// ---------------------------------------------------------------------------
// Kernel 1: xin[(b*T+t), h] = x[(b*T+t), i] * w_in[h, i] + b_in[h]
// Written into the hidden_list region of d_out (row-major [32768, 1024]).
// Each wave: one 16-row M-tile x four 16-col N-tiles (A-frag reused 4x).
// ---------------------------------------------------------------------------
__global__ void __launch_bounds__(256) k_inproj(const float* __restrict__ x,
                                                const float* __restrict__ w,
                                                const float* __restrict__ bias,
                                                float* __restrict__ out) {
  const int lane = threadIdx.x & 31;
  const int wave = threadIdx.x >> 5;
  const int tile = blockIdx.x * 8 + wave;
  const int mt = tile >> 4;            // 2048 M-tiles
  const int ng = tile & 15;            // 16 groups of 4 N-tiles
  const int m0 = mt * 16;
  const int nbase = ng * 64;
  const int half = lane >> 4, l15 = lane & 15;

  FragC c[4];
#pragma unroll
  for (int j = 0; j < 4; ++j)
#pragma unroll
    for (int i = 0; i < 8; ++i) c[j].f[i] = 0.f;

  const float* arow = x + (size_t)(m0 + l15) * NI;
  const float* brow[4];
#pragma unroll
  for (int j = 0; j < 4; ++j) brow[j] = w + (size_t)(nbase + j * 16 + l15) * NI;

#pragma unroll
  for (int kc = 0; kc < NI / 32; ++kc) {
    const int k0 = kc * 32;
    Frag16 a;
    loadB_f32(a, arow + k0 + (half << 4));
    const int kb = k0 + (half << 4);
#pragma unroll
    for (int j = 0; j < 4; ++j) {
      Frag16 b;
      loadB_f32(b, brow[j] + kb);
      c[j].v = __builtin_amdgcn_wmma_f32_16x16x32_bf16(false, a.v, false, b.v,
                                                       (short)0, c[j].v, false, false);
    }
  }

#pragma unroll
  for (int j = 0; j < 4; ++j) {
    const int col = nbase + j * 16 + l15;
    const float bb = bias[col];
#pragma unroll
    for (int r = 0; r < 8; ++r) {
      const int m = m0 + r + (half << 3);
      out[(size_t)m * NH + col] = c[j].f[r] + bb;
    }
  }
}

// ---------------------------------------------------------------------------
// Kernel 2: persistent recurrence. 32 blocks x 256 threads (8 waves).
// Block owns 32 H-columns (bf16 W_hh slice in 64KB LDS); wave owns 16 B-rows
// and keeps 2 accumulators (2 WMMAs per A-load, independent D-chains).
// tanh(h) double-buffered bf16; h fp32 in workspace; hlist holds xin -> h_new.
// ---------------------------------------------------------------------------
__global__ void __launch_bounds__(256) k_recur(const float* __restrict__ hidden0,
                                               const float* __restrict__ whh,
                                               const float* __restrict__ bhh,
                                               const float* __restrict__ noise,
                                               float* __restrict__ hlist,
                                               float* __restrict__ hfin,
                                               float* __restrict__ hf,
                                               unsigned short* __restrict__ ht,
                                               unsigned int* __restrict__ ctr) {
  __shared__ unsigned short ldsw[32 * NH];   // 64 KB: this block's 32 W_hh columns, bf16

  const int lane = threadIdx.x & 31;
  const int wave = threadIdx.x >> 5;
  const int half = lane >> 4, l15 = lane & 15;
  const int n0 = blockIdx.x * 32;            // column slice
  const int m0 = wave * 16;                  // row slice (B = 8 waves * 16)
  const unsigned int nb = gridDim.x;

  // Stage W_hh[n0..n0+31, :] -> LDS bf16 (packed-pair converts, read once).
  {
    unsigned int* ldsu = reinterpret_cast<unsigned int*>(ldsw);
    for (int i = threadIdx.x; i < 32 * NH / 2; i += 256) {
      const int row = i >> 9;              // 512 uint pairs per row
      const int k2 = (i & 511) * 2;
      const float* src = whh + (size_t)(n0 + row) * NH + k2;
      ldsu[i] = pack2bf(src[0], src[1]);
    }
  }

  // Init fp32 state + tanh buffer (grid-strided slice: 8192 threads x 16 elems).
  {
    const int base = (blockIdx.x * 256 + threadIdx.x) * 16;
#pragma unroll
    for (int i = 0; i < 16; ++i) {
      const float h0 = hidden0[base + i];
      hf[base + i] = h0;
      ht[base + i] = f2bf(TANHF(h0));
    }
  }
  gridbar(ctr, nb);   // epoch 0: init + LDS visible

  const unsigned short* bcol0 = &ldsw[(l15)      * NH];
  const unsigned short* bcol1 = &ldsw[(16 + l15) * NH];

  for (int t = 0; t < TT; ++t) {
    const unsigned short* htc = ht + (size_t)(t & 1) * (BB * NH);
    unsigned short*       htn = ht + (size_t)((t + 1) & 1) * (BB * NH);

    FragC c0, c1;
#pragma unroll
    for (int i = 0; i < 8; ++i) { c0.f[i] = 0.f; c1.f[i] = 0.f; }

    const unsigned short* arow = htc + (size_t)(m0 + l15) * NH;

#pragma unroll 4
    for (int kc = 0; kc < NH / 32; ++kc) {
      const int k0 = kc * 32;
      const int ks = k0 + (half << 4);
      Frag16 a, b0, b1;
      a.q[0] = *reinterpret_cast<const u32x4*>(arow + ks);
      a.q[1] = *reinterpret_cast<const u32x4*>(arow + ks + 8);
      __builtin_prefetch(arow + ks + 64, 0, 0);     // global_prefetch_b8 on the h stream
      const int kb = k0 + (half << 4);
      b0.q[0] = *reinterpret_cast<const u32x4*>(bcol0 + kb);      // ds_load_b128
      b0.q[1] = *reinterpret_cast<const u32x4*>(bcol0 + kb + 8);
      b1.q[0] = *reinterpret_cast<const u32x4*>(bcol1 + kb);
      b1.q[1] = *reinterpret_cast<const u32x4*>(bcol1 + kb + 8);
      c0.v = __builtin_amdgcn_wmma_f32_16x16x32_bf16(false, a.v, false, b0.v,
                                                     (short)0, c0.v, false, false);
      c1.v = __builtin_amdgcn_wmma_f32_16x16x32_bf16(false, a.v, false, b1.v,
                                                     (short)0, c1.v, false, false);
    }

    // Leaky-integrator update in fp32 for the owned 16x32 tile.
#pragma unroll
    for (int j = 0; j < 2; ++j) {
      FragC& c = j ? c1 : c0;
      const int col = n0 + j * 16 + l15;
      const float bh = bhh[col];
#pragma unroll
      for (int r = 0; r < 8; ++r) {
        const int bi = m0 + r + (half << 3);
        const size_t hidx = (size_t)bi * NH + col;
        const size_t oidx = ((size_t)bi * TT + t) * NH + col;
        const float hold = hf[hidx];
        const float xin  = hlist[oidx];                               // x@W_in + b_in
        const float nz   = noise[((size_t)t * BB + bi) * NH + col];
        const float tmp  = c.f[r] + xin + bh;
        const float hnew = (1.f - ALPHA_C) * hold + ALPHA_C * tmp + nz * NOISE_STD;
        hf[hidx]    = hnew;
        hlist[oidx] = hnew;                                           // hidden_list[b,t,:]
        htn[hidx]   = f2bf(TANHF(hnew));
      }
    }

    gridbar(ctr, nb * (unsigned)(t + 2));
  }

  // h_final = hf (all writes visible after last barrier)
  {
    const int base = (blockIdx.x * 256 + threadIdx.x) * 16;
#pragma unroll
    for (int i = 0; i < 16; ++i) hfin[base + i] = hf[base + i];
  }
}

// ---------------------------------------------------------------------------
// Kernel 3: output_list[(b*T+t), o] = hidden_list[(b*T+t), h] * w_out[o, h] + b_out
// Each wave handles one M-tile x all 4 N-tiles: hidden_list read/converted once.
// ---------------------------------------------------------------------------
__global__ void __launch_bounds__(256) k_outproj(const float* __restrict__ hl,
                                                 const float* __restrict__ w,
                                                 const float* __restrict__ bias,
                                                 float* __restrict__ out) {
  const int lane = threadIdx.x & 31;
  const int wave = threadIdx.x >> 5;
  const int mt = blockIdx.x * 8 + wave;   // 2048 M-tiles
  const int m0 = mt * 16;
  const int half = lane >> 4, l15 = lane & 15;

  FragC c[4];
#pragma unroll
  for (int j = 0; j < 4; ++j)
#pragma unroll
    for (int i = 0; i < 8; ++i) c[j].f[i] = 0.f;

  const float* arow = hl + (size_t)(m0 + l15) * NH;
  const float* brow[4];
#pragma unroll
  for (int j = 0; j < 4; ++j) brow[j] = w + (size_t)(j * 16 + l15) * NH;

#pragma unroll 2
  for (int kc = 0; kc < NH / 32; ++kc) {
    const int k0 = kc * 32;
    Frag16 a;
    loadB_f32(a, arow + k0 + (half << 4));
    const int kb = k0 + (half << 4);
#pragma unroll
    for (int j = 0; j < 4; ++j) {
      Frag16 b;
      loadB_f32(b, brow[j] + kb);
      c[j].v = __builtin_amdgcn_wmma_f32_16x16x32_bf16(false, a.v, false, b.v,
                                                       (short)0, c[j].v, false, false);
    }
  }

#pragma unroll
  for (int j = 0; j < 4; ++j) {
    const int col = j * 16 + l15;
    const float bb = bias[col];
#pragma unroll
    for (int r = 0; r < 8; ++r) {
      const int m = m0 + r + (half << 3);
      out[(size_t)m * NO + col] = c[j].f[r] + bb;
    }
  }
}

// ---------------------------------------------------------------------------
extern "C" void kernel_launch(void* const* d_in, const int* in_sizes, int n_in,
                              void* d_out, int out_size, void* d_ws, size_t ws_size,
                              hipStream_t stream) {
  const float* x     = (const float*)d_in[0];   // [B,T,NI]
  const float* h0    = (const float*)d_in[1];   // [B,NH]
  const float* winw  = (const float*)d_in[2];   // [NH,NI]
  const float* winb  = (const float*)d_in[3];   // [NH]
  const float* whhw  = (const float*)d_in[4];   // [NH,NH]
  const float* whhb  = (const float*)d_in[5];   // [NH]
  const float* woutw = (const float*)d_in[6];   // [NO,NH]
  const float* woutb = (const float*)d_in[7];   // [NO]
  const float* noise = (const float*)d_in[8];   // [T,B,NH]

  float* out   = (float*)d_out;
  float* hlist = out;                            // [B,T,NH]
  float* olist = out + (size_t)BB * TT * NH;     // [B,T,NO]
  float* hfin  = olist + (size_t)BB * TT * NO;   // [B,NH]

  char* ws = (char*)d_ws;
  unsigned int*   ctr = (unsigned int*)ws;                                     // barrier counter
  float*          hf  = (float*)(ws + 256);                                    // fp32 h state
  unsigned short* ht  = (unsigned short*)(ws + 256 + sizeof(float) * BB * NH); // bf16 tanh(h) x2

  hipMemsetAsync(ctr, 0, sizeof(unsigned int), stream);

  // 1) xin -> hidden_list slots of d_out
  k_inproj<<<(BB * TT / 16) * 16 / 8, 256, 0, stream>>>(x, winw, winb, hlist);
  // 2) persistent recurrence (32 blocks; one 32-col slice per block)
  k_recur<<<NH / 32, 256, 0, stream>>>(h0, whhw, whhb, noise, hlist, hfin, hf, ht, ctr);
  // 3) output projection
  k_outproj<<<(BB * TT / 16) / 8, 256, 0, stream>>>(hlist, woutw, woutb, olist);
}
